// VQVAE_37391985279342
// MI455X (gfx1250) — matrix operand order repaired and checked by reading, not compiled, not run on hardware
//
#include <hip/hip_runtime.h>

// ---------------------------------------------------------------------------
// Types for CDNA5 WMMA (wave32): bf16 16x16x32, f32 accumulate
// ---------------------------------------------------------------------------
typedef __attribute__((ext_vector_type(16))) __bf16 v16bf;
typedef __attribute__((ext_vector_type(8)))  __bf16 v8bf;
typedef __attribute__((ext_vector_type(8)))  float  v8f;

// CDNA5 async global->LDS DMA (ASYNCcnt-tracked, no VGPR round trip).
__device__ __forceinline__ void async_b128(unsigned ldsOff, const void* gaddr) {
  asm volatile("global_load_async_to_lds_b128 %0, %1, off"
               :: "v"(ldsOff), "v"(gaddr) : "memory");
}
__device__ __forceinline__ void wait_async0() {
  asm volatile("s_wait_asynccnt 0" ::: "memory");
}

// ---------------------------------------------------------------------------
// Implicit-GEMM convolution / deconvolution on WMMA.
//   C[M x N] = A[M x K] * B[K x N]
//   A = reordered weights, bf16, row-major [co][K], K = taps*Cin (ci fastest)
//   B = gathered on the fly from NHWC bf16 activations (implicit im2col)
//   mode 0: conv  (stride/pad/KW),  mode 1: convT parity class (pb,pc)
// Tiling: 8 waves/block; wave tile 32Mx32N; block tile 128M x 64N; K step 32.
// Both operands double-buffered in LDS via global_load_async_to_lds_b128;
// LDS row stride 40 halfs (20 banks) -> conflict-free ds_load_b128 frags.
// ---------------------------------------------------------------------------
struct GP {
  const __bf16* A;
  const __bf16* X;
  float*        Y;
  const float*  bias;     // [M] or null
  int M, N, K;
  int IH, IW, Cin, cinShift;
  int OH, OW;             // per-launch output grid (per-parity for convT)
  int KW, stride, pad;    // conv mode
  int mode, pb, pc;       // convT parity
  int OHt, OWt;           // convT full output dims
  int CoutStride;         // channel stride of Y
};

#define LDS_STRIDE 40  // halfs per K-row slice (32 data + 8 pad = 20 banks)

__global__ __launch_bounds__(256) void wmma_conv_gemm(GP p) {
  __shared__ __bf16 As[2][128 * LDS_STRIDE];
  __shared__ __bf16 Bs[2][64 * LDS_STRIDE];

  const int tid  = threadIdx.x;
  const int lane = tid & 31;
  const int wave = tid >> 5;
  const int lm   = lane & 15;
  const int half = lane >> 4;
  const int waveM = wave >> 1;   // 0..3
  const int waveN = wave & 1;    // 0..1
  const int mBlk = blockIdx.x * 128;
  const int nBlk = blockIdx.y * 64;
  const int mW = mBlk + waveM * 32;
  const int nW = nBlk + waveN * 32;

  // ---- staging assignment (loop-invariant) ----
  // A tile: 128 rows x 32 halfs; thread t covers 32B: row = t&127, chunk = t>>7.
  const int sAr = tid & 127;
  const int sAh = tid >> 7;
  const int aRowRaw = mBlk + sAr;
  const int aRow = (aRowRaw < p.M) ? aRowRaw : (p.M - 1);   // clamp: no branches in loop
  const __bf16* aSrc = p.A + (long long)aRow * p.K + sAh * 16;
  const int aDst = sAr * LDS_STRIDE + sAh * 16;
  // B tile: 64 pixels x 32 halfs; threads 0..127: pixel = t>>1, k-chunk = t&1.
  const bool doB = tid < 128;
  const int bN = tid >> 1;
  const int bH = tid & 1;
  int bBat = 0, bPy = 0, bPx = 0;
  bool bNOK = false;
  {
    int n = nBlk + bN;
    bNOK = (n < p.N);
    int nn = bNOK ? n : 0;
    bPx = nn % p.OW;
    int t = nn / p.OW;
    bPy = t % p.OH;
    bBat = t / p.OH;
  }

  auto stage = [&](int buf, int k0) {
    {   // A: addresses always valid -> pure async DMA
      unsigned d = (unsigned)(size_t)&As[buf][aDst];
      const __bf16* g = aSrc + k0;
      async_b128(d, g);
      async_b128(d + 16, g + 8);
    }
    if (doB) {  // B: implicit im2col gather, async for valid taps, zeros for padding
      int k16 = k0 + bH * 16;
      int tap = k16 >> p.cinShift;
      int ci  = k16 & (p.Cin - 1);
      int iy, ix;
      if (p.mode == 0) {
        int ky = tap / p.KW;
        int kx = tap - ky * p.KW;
        iy = bPy * p.stride - p.pad + ky;
        ix = bPx * p.stride - p.pad + kx;
      } else {
        iy = bPy + p.pb - (tap >> 1);
        ix = bPx + p.pc - (tap & 1);
      }
      __bf16* dst = &Bs[buf][bN * LDS_STRIDE + bH * 16];
      bool ok = bNOK & ((unsigned)iy < (unsigned)p.IH) & ((unsigned)ix < (unsigned)p.IW);
      if (ok) {
        const __bf16* g = p.X +
            (((long long)bBat * p.IH + iy) * p.IW + ix) * (long long)p.Cin + ci;
        unsigned d = (unsigned)(size_t)dst;
        async_b128(d, g);
        async_b128(d + 16, g + 8);
      } else {
        v8bf z;
#pragma unroll
        for (int e = 0; e < 8; ++e) z[e] = (__bf16)0.0f;
        *(v8bf*)dst = z;
        *(v8bf*)(dst + 8) = z;
      }
    }
  };

  // ---- per-lane output-pixel decode for the epilogue ----
  int batc[2], py[2], px[2];
  long long nIdx[2];
  bool nOK[2];
#pragma unroll
  for (int f = 0; f < 2; ++f) {
    int n = nW + f * 16 + lm;
    nIdx[f] = n;
    nOK[f]  = (n < p.N);
    int nn  = nOK[f] ? n : 0;
    px[f]   = nn % p.OW;
    int t   = nn / p.OW;
    py[f]   = t % p.OH;
    batc[f] = t / p.OH;
  }

  v8f acc[2][2];
#pragma unroll
  for (int a = 0; a < 2; ++a)
#pragma unroll
    for (int b = 0; b < 2; ++b)
#pragma unroll
      for (int r = 0; r < 8; ++r) acc[a][b][r] = 0.0f;

  const bool mOK[2] = { mW < p.M, (mW + 16) < p.M };

  // ---- software pipeline: prefetch tile 0, then overlap DMA with WMMA ----
  stage(0, 0);
  wait_async0();
  __syncthreads();

  for (int k0 = 0; k0 < p.K; k0 += 32) {
    const int buf = (k0 >> 5) & 1;
    if (k0 + 32 < p.K) stage(buf ^ 1, k0 + 32);

    // A frags: 16-bit A 16x32 layout: lane m = lm, k = 16*(e>=8) + half*8 + (e&7)
    v16bf afrag[2];
#pragma unroll
    for (int mf = 0; mf < 2; ++mf) {
      const __bf16* s = &As[buf][(waveM * 32 + mf * 16 + lm) * LDS_STRIDE + half * 8];
      v8bf lo = *(const v8bf*)s;         // K = k0 + half*8 .. +8
      v8bf hi = *(const v8bf*)(s + 16);  // K = k0 + 16 + half*8 .. +8
#pragma unroll
      for (int e = 0; e < 8; ++e) { afrag[mf][e] = lo[e]; afrag[mf][e + 8] = hi[e]; }
    }
    // B frags: same layout as A: k = 16*(e>=8) + half*8 + (e&7)
    v16bf bfrag[2];
#pragma unroll
    for (int nf = 0; nf < 2; ++nf) {
      const __bf16* s = &Bs[buf][(waveN * 32 + nf * 16 + lm) * LDS_STRIDE + half * 8];
      v8bf lo = *(const v8bf*)s;
      v8bf hi = *(const v8bf*)(s + 16);
#pragma unroll
      for (int e = 0; e < 8; ++e) { bfrag[nf][e] = lo[e]; bfrag[nf][e + 8] = hi[e]; }
    }

#pragma unroll
    for (int mf = 0; mf < 2; ++mf)
#pragma unroll
      for (int nf = 0; nf < 2; ++nf)
        acc[mf][nf] = __builtin_amdgcn_wmma_f32_16x16x32_bf16(
            false, afrag[mf], false, bfrag[nf], (short)0, acc[mf][nf], false, false);

    wait_async0();     // next tile's DMA complete (compiler can't track ASYNCcnt)
    __syncthreads();   // all waves done reading current buffers
  }

  // ---- Epilogue: C/D layout: n = lm, m = r + 8*half. 2x float4 stores.
#pragma unroll
  for (int mf = 0; mf < 2; ++mf) {
    if (!mOK[mf]) continue;
    int m0 = mW + mf * 16 + 8 * half;
    float bia[8];
#pragma unroll
    for (int r = 0; r < 8; ++r) bia[r] = p.bias ? p.bias[m0 + r] : 0.0f;
#pragma unroll
    for (int nf = 0; nf < 2; ++nf) {
      if (!nOK[nf]) continue;
      long long pix;
      if (p.mode == 0) {
        pix = nIdx[nf];
      } else {
        int oy = 2 * py[nf] + p.pb;
        int ox = 2 * px[nf] + p.pc;
        pix = ((long long)batc[nf] * p.OHt + oy) * p.OWt + ox;
      }
      float* yp = p.Y + pix * (long long)p.CoutStride + m0;
      float4 s0 = make_float4(acc[mf][nf][0] + bia[0], acc[mf][nf][1] + bia[1],
                              acc[mf][nf][2] + bia[2], acc[mf][nf][3] + bia[3]);
      float4 s1 = make_float4(acc[mf][nf][4] + bia[4], acc[mf][nf][5] + bia[5],
                              acc[mf][nf][6] + bia[6], acc[mf][nf][7] + bia[7]);
      *(float4*)(yp)     = s0;
      *(float4*)(yp + 4) = s1;
    }
  }
}

// ---------------------------------------------------------------------------
// BatchNorm: deterministic two-level reduction (partials -> tree reduce)
// ---------------------------------------------------------------------------
#define PPB 4096  // pixels per stats block

__global__ __launch_bounds__(256) void bn_stats_kernel(const float* __restrict__ x,
    float* __restrict__ ps, float* __restrict__ pq, int Cstride, int chunks) {
  const int c = blockIdx.x;
  const int chunk = blockIdx.y;
  long long p0 = (long long)chunk * PPB;
  float s = 0.f, q = 0.f;
  for (int i = threadIdx.x; i < PPB; i += 256) {
    float v = x[(p0 + i) * (long long)Cstride + c];
    s += v; q += v * v;
  }
  __shared__ float ls[256], lq[256];
  ls[threadIdx.x] = s; lq[threadIdx.x] = q;
  __syncthreads();
  for (int st = 128; st > 0; st >>= 1) {
    if (threadIdx.x < st) { ls[threadIdx.x] += ls[threadIdx.x + st]; lq[threadIdx.x] += lq[threadIdx.x + st]; }
    __syncthreads();
  }
  if (threadIdx.x == 0) {
    ps[(long long)c * chunks + chunk] = ls[0];
    pq[(long long)c * chunks + chunk] = lq[0];
  }
}

__global__ __launch_bounds__(256) void bn_reduce_kernel(const float* __restrict__ ps,
    const float* __restrict__ pq, float* __restrict__ sum, float* __restrict__ sumsq, int chunks) {
  const int c = blockIdx.x;
  float s = 0.f, q = 0.f;
  for (int i = threadIdx.x; i < chunks; i += 256) {
    s += ps[(long long)c * chunks + i];
    q += pq[(long long)c * chunks + i];
  }
  __shared__ float ls[256], lq[256];
  ls[threadIdx.x] = s; lq[threadIdx.x] = q;
  __syncthreads();
  for (int st = 128; st > 0; st >>= 1) {
    if (threadIdx.x < st) { ls[threadIdx.x] += ls[threadIdx.x + st]; lq[threadIdx.x] += lq[threadIdx.x + st]; }
    __syncthreads();
  }
  if (threadIdx.x == 0) { sum[c] = ls[0]; sumsq[c] = lq[0]; }
}

__global__ __launch_bounds__(256) void bn_apply_kernel(const float* __restrict__ x,
    const float* __restrict__ sum, const float* __restrict__ sumsq,
    const float* __restrict__ g, const float* __restrict__ bta,
    __bf16* __restrict__ outb, float* __restrict__ outf,
    long long total, int C, float invP, int relu) {
  long long stride = (long long)gridDim.x * blockDim.x;
  for (long long i = (long long)blockIdx.x * blockDim.x + threadIdx.x; i < total; i += stride) {
    long long pix = i / C;
    int c = (int)(i - pix * C);
    float mu  = sum[c] * invP;
    float var = sumsq[c] * invP - mu * mu;
    float v = (x[i] - mu) * rsqrtf(var + 1e-5f) * g[c] + bta[c];
    if (relu) v = fmaxf(v, 0.f);
    if (outb) outb[i] = (__bf16)v;
    if (outf) outf[i] = v;
  }
}

// ---------------------------------------------------------------------------
// Elementwise helpers
// ---------------------------------------------------------------------------
__global__ __launch_bounds__(256) void add_inplace_kernel(float* __restrict__ acc,
    const float* __restrict__ other, long long n) {
  long long stride = (long long)gridDim.x * blockDim.x;
  for (long long i = (long long)blockIdx.x * blockDim.x + threadIdx.x; i < n; i += stride)
    acc[i] += other[i];
}

__global__ __launch_bounds__(256) void pack_image_kernel(const float* __restrict__ img,
    __bf16* __restrict__ xp, int H, int W) {
  long long total = (long long)16 * H * W * 16;
  long long stride = (long long)gridDim.x * blockDim.x;
  for (long long i = (long long)blockIdx.x * blockDim.x + threadIdx.x; i < total; i += stride) {
    int c = (int)(i & 15);
    long long r = i >> 4;
    int x = (int)(r % W); r /= W;
    int y = (int)(r % H); int b = (int)(r / H);
    float v = (c < 3) ? img[(((long long)b * 3 + c) * H + y) * W + x] : 0.f;
    xp[i] = (__bf16)v;
  }
}

// weights OIHW f32 -> [co][ky][kx][ci] bf16, ci padded to CinP
__global__ __launch_bounds__(256) void reorder_conv_w_kernel(const float* __restrict__ w,
    __bf16* __restrict__ out, int Cout, int Cin, int CinP, int KH, int KW) {
  long long K = (long long)KH * KW * CinP;
  long long total = Cout * K;
  long long stride = (long long)gridDim.x * blockDim.x;
  for (long long i = (long long)blockIdx.x * blockDim.x + threadIdx.x; i < total; i += stride) {
    long long co = i / K;
    long long k  = i - co * K;
    int tap = (int)(k / CinP);
    int ci  = (int)(k - (long long)tap * CinP);
    int ky = tap / KW, kx = tap - ky * KW;
    float v = (ci < Cin) ? w[((co * Cin + ci) * KH + ky) * KW + kx] : 0.f;
    out[i] = (__bf16)v;
  }
}

// convT weights (Cout,Cin,4,4) f32 -> 4 parity slices [p][co][t][ci] bf16 (co padded to Mpad)
__global__ __launch_bounds__(256) void reorder_convt_w_kernel(const float* __restrict__ w,
    __bf16* __restrict__ out, int Cout, int Mpad, int Cin) {
  long long K4 = 4LL * Cin;
  long long total = 4LL * Mpad * K4;
  long long stride = (long long)gridDim.x * blockDim.x;
  for (long long i = (long long)blockIdx.x * blockDim.x + threadIdx.x; i < total; i += stride) {
    long long pslice = i / (Mpad * K4);
    long long rem = i - pslice * Mpad * K4;
    long long co = rem / K4;
    long long k = rem - co * K4;
    int t = (int)(k / Cin);
    int ci = (int)(k - (long long)t * Cin);
    int pb = (int)(pslice >> 1), pc = (int)(pslice & 1);
    int ky = (1 - pb) + 2 * (t >> 1);
    int kx = (1 - pc) + 2 * (t & 1);
    float v = (co < Cout) ? w[((co * Cin + ci) * 4 + ky) * 4 + kx] : 0.f;
    out[i] = (__bf16)v;
  }
}

__global__ __launch_bounds__(256) void cast_bf16_kernel(const float* __restrict__ in,
    __bf16* __restrict__ out, long long n) {
  long long stride = (long long)gridDim.x * blockDim.x;
  for (long long i = (long long)blockIdx.x * blockDim.x + threadIdx.x; i < n; i += stride)
    out[i] = (__bf16)in[i];
}

__global__ __launch_bounds__(256) void pad_bias_kernel(const float* __restrict__ b,
    float* __restrict__ out, int n, int npad) {
  int i = blockIdx.x * blockDim.x + threadIdx.x;
  if (i < npad) out[i] = (i < n) ? b[i] : 0.f;
}

// ---------------------------------------------------------------------------
// Vector quantization: ||e||^2, argmin(||e||^2 - 2*score), gather
// ---------------------------------------------------------------------------
__global__ __launch_bounds__(256) void enorm_kernel(const float* __restrict__ emb,
    float* __restrict__ enorm, int D) {
  const int e = blockIdx.x;
  float s = 0.f;
  for (int c = threadIdx.x; c < D; c += 256) {
    float v = emb[(long long)e * D + c];
    s += v * v;
  }
  __shared__ float ls[256];
  ls[threadIdx.x] = s;
  __syncthreads();
  for (int st = 128; st > 0; st >>= 1) {
    if (threadIdx.x < st) ls[threadIdx.x] += ls[threadIdx.x + st];
    __syncthreads();
  }
  if (threadIdx.x == 0) enorm[e] = ls[0];
}

__global__ __launch_bounds__(256) void argmin_kernel(const float* __restrict__ scores,
    const float* __restrict__ enorm, int* __restrict__ idx, int NE) {
  const int pix = blockIdx.x;
  float best = 3.4e38f;
  int bi = 0x7fffffff;
  for (int e = threadIdx.x; e < NE; e += 256) {
    float d = enorm[e] - 2.0f * scores[(long long)pix * NE + e];
    if (d < best) { best = d; bi = e; }
  }
  __shared__ float bs[256];
  __shared__ int   bx[256];
  bs[threadIdx.x] = best; bx[threadIdx.x] = bi;
  __syncthreads();
  for (int st = 128; st > 0; st >>= 1) {
    if (threadIdx.x < st) {
      float o = bs[threadIdx.x + st];
      int oi = bx[threadIdx.x + st];
      if (o < bs[threadIdx.x] || (o == bs[threadIdx.x] && oi < bx[threadIdx.x])) {
        bs[threadIdx.x] = o; bx[threadIdx.x] = oi;
      }
    }
    __syncthreads();
  }
  if (threadIdx.x == 0) idx[pix] = bx[0];
}

__global__ __launch_bounds__(256) void gather_kernel(const float* __restrict__ emb,
    const int* __restrict__ idx, __bf16* __restrict__ zqb, float* __restrict__ zqf,
    long long P, int D) {
  long long total = P * D;
  long long stride = (long long)gridDim.x * blockDim.x;
  for (long long i = (long long)blockIdx.x * blockDim.x + threadIdx.x; i < total; i += stride) {
    long long pix = i / D;
    int c = (int)(i - pix * D);
    float v = emb[(long long)idx[pix] * D + c];
    zqb[i] = (__bf16)v;
    zqf[i] = v;
  }
}

// ---------------------------------------------------------------------------
// Final: BN (stats already reduced, C=3, channel stride 16) + tanh -> NCHW f32
// ---------------------------------------------------------------------------
__global__ __launch_bounds__(256) void final_out_kernel(const float* __restrict__ yf,
    const float* __restrict__ sum, const float* __restrict__ sumsq,
    const float* __restrict__ g, const float* __restrict__ bta,
    float* __restrict__ out, int H, int W, float invP) {
  long long total = (long long)16 * 3 * H * W;
  long long stride = (long long)gridDim.x * blockDim.x;
  for (long long i = (long long)blockIdx.x * blockDim.x + threadIdx.x; i < total; i += stride) {
    int x = (int)(i % W);
    long long t = i / W;
    int y = (int)(t % H); t /= H;
    int c = (int)(t % 3);
    int b = (int)(t / 3);
    float v = yf[(((long long)b * H + y) * W + x) * 16 + c];
    float mu  = sum[c] * invP;
    float var = sumsq[c] * invP - mu * mu;
    float r = (v - mu) * rsqrtf(var + 1e-5f) * g[c] + bta[c];
    out[i] = tanhf(r);
  }
}

// ---------------------------------------------------------------------------
// Host orchestration
// ---------------------------------------------------------------------------
static unsigned gsb(long long n) {
  long long b = (n + 255) / 256;
  return (unsigned)(b > 32768 ? 32768 : (b < 1 ? 1 : b));
}

extern "C" void kernel_launch(void* const* d_in, const int* in_sizes, int n_in,
                              void* d_out, int out_size, void* d_ws, size_t ws_size,
                              hipStream_t stream) {
  (void)in_sizes; (void)n_in; (void)out_size; (void)ws_size;
  const float* ori     = (const float*)d_in[0];
  const float* e_c1_w  = (const float*)d_in[1];
  const float* e_bn1_g = (const float*)d_in[2];
  const float* e_bn1_b = (const float*)d_in[3];
  const float* e_c2_w  = (const float*)d_in[4];
  const float* e_bn2_g = (const float*)d_in[5];
  const float* e_bn2_b = (const float*)d_in[6];
  const float* e_r1_w  = (const float*)d_in[7];
  const float* e_rbn1_g= (const float*)d_in[8];
  const float* e_rbn1_b= (const float*)d_in[9];
  const float* e_r2_w  = (const float*)d_in[10];
  const float* e_r2_b  = (const float*)d_in[11];
  const float* e_rbn2_g= (const float*)d_in[12];
  const float* e_rbn2_b= (const float*)d_in[13];
  const float* e_bn3_g = (const float*)d_in[14];
  const float* e_bn3_b = (const float*)d_in[15];
  const float* emb     = (const float*)d_in[16];
  const float* d_r1_w  = (const float*)d_in[17];
  const float* d_rbn1_g= (const float*)d_in[18];
  const float* d_rbn1_b= (const float*)d_in[19];
  const float* d_r2_w  = (const float*)d_in[20];
  const float* d_rbn2_g= (const float*)d_in[21];
  const float* d_rbn2_b= (const float*)d_in[22];
  const float* d_bn0_g = (const float*)d_in[23];
  const float* d_bn0_b = (const float*)d_in[24];
  const float* d_t1_w  = (const float*)d_in[25];
  const float* d_bn1_g = (const float*)d_in[26];
  const float* d_bn1_b = (const float*)d_in[27];
  const float* d_t2_w  = (const float*)d_in[28];
  const float* d_t2_b  = (const float*)d_in[29];
  const float* d_bn2_g = (const float*)d_in[30];
  const float* d_bn2_b = (const float*)d_in[31];
  float* out = (float*)d_out;

  // ---- workspace layout ----
  char* ws = (char*)d_ws;
  size_t off = 0;
  auto alloc = [&](size_t bytes) -> void* {
    void* p = ws + off;
    off = (off + bytes + 255) & ~(size_t)255;
    return p;
  };
  const long long P128 = 16LL * 128 * 128;   // 262144
  const long long P64  = 16LL * 64 * 64;     // 65536
  const long long P256 = 16LL * 256 * 256;   // 1048576

  __bf16* XP    = (__bf16*)alloc(16LL * 256 * 256 * 16 * 2);
  __bf16* W_C1  = (__bf16*)alloc(512LL * 256 * 2);
  __bf16* W_C2  = (__bf16*)alloc(512LL * 8192 * 2);
  __bf16* W_ER1 = (__bf16*)alloc(512LL * 4608 * 2);
  __bf16* W_ER2 = (__bf16*)alloc(512LL * 512 * 2);
  __bf16* W_EMB = (__bf16*)alloc(512LL * 512 * 2);
  __bf16* W_DR1 = (__bf16*)alloc(512LL * 4608 * 2);
  __bf16* W_DR2 = (__bf16*)alloc(512LL * 512 * 2);
  __bf16* W_T1  = (__bf16*)alloc(4LL * 512 * 2048 * 2);
  __bf16* W_T2  = (__bf16*)alloc(4LL * 16 * 2048 * 2);
  float*  BT2   = (float*)alloc(16 * 4);
  float*  SUMB  = (float*)alloc(512 * 4);
  float*  SQB   = (float*)alloc(512 * 4);
  float*  PSUM  = (float*)alloc(512LL * 256 * 4);
  float*  PSQ   = (float*)alloc(512LL * 256 * 4);
  float*  ENORM = (float*)alloc(512 * 4);
  int*    IDX   = (int*)alloc(P64 * 4);
  float*  BIG1  = (float*)alloc(P128 * 512 * 4);   // conv1 out / convT1 out
  __bf16* BIG2  = (__bf16*)alloc(P128 * 512 * 2);  // bn1 out / bn1d out
  float*  M64_1 = (float*)alloc(P64 * 512 * 4);    // conv2 out / scores / d_r2 out
  float*  M64_2 = (float*)alloc(P64 * 512 * 4);    // X2f / z_e sum
  float*  M64_3 = (float*)alloc(P64 * 512 * 4);    // resconv out / r3f
  float*  M64_4 = (float*)alloc(P64 * 512 * 4);    // e_r2 out / zqf
  __bf16* M64B1 = (__bf16*)alloc(P64 * 512 * 2);   // X2b / zqb
  __bf16* M64B2 = (__bf16*)alloc(P64 * 512 * 2);   // resblock bf16
  __bf16* M64B3 = (__bf16*)alloc(P64 * 512 * 2);   // z_e bf16 / bn0 out
  float*  YF    = (float*)alloc(16LL * 256 * 256 * 16 * 4);

  // ---- GEMM launchers ----
  auto launch_conv = [&](const __bf16* A, const __bf16* X, float* Y, const float* bias,
                         int M, int IH, int IW, int Cin, int OH, int OW,
                         int KH, int KW, int st, int pd, int CoutStride) {
    GP p{};
    p.A = A; p.X = X; p.Y = Y; p.bias = bias;
    p.M = M; p.N = 16 * OH * OW; p.K = KH * KW * Cin;
    p.IH = IH; p.IW = IW; p.Cin = Cin; p.cinShift = (Cin == 512) ? 9 : 4;
    p.OH = OH; p.OW = OW; p.KW = KW; p.stride = st; p.pad = pd;
    p.mode = 0; p.pb = 0; p.pc = 0; p.OHt = OH; p.OWt = OW;
    p.CoutStride = CoutStride;
    dim3 g((M + 127) / 128, (p.N + 63) / 64);
    wmma_conv_gemm<<<g, 256, 0, stream>>>(p);
  };
  auto launch_convt = [&](const __bf16* Wp, const __bf16* X, float* Y, const float* bias,
                          int M, int Mpad, int IH, int IW, int Cin) {
    for (int pb = 0; pb < 2; ++pb)
      for (int pc = 0; pc < 2; ++pc) {
        GP p{};
        p.A = Wp + (long long)(pb * 2 + pc) * Mpad * 4 * Cin;
        p.X = X; p.Y = Y; p.bias = bias;
        p.M = Mpad; p.N = 16 * IH * IW; p.K = 4 * Cin;
        p.IH = IH; p.IW = IW; p.Cin = Cin; p.cinShift = 9;
        p.OH = IH; p.OW = IW; p.KW = 2; p.stride = 1; p.pad = 0;
        p.mode = 1; p.pb = pb; p.pc = pc; p.OHt = 2 * IH; p.OWt = 2 * IW;
        p.CoutStride = Mpad;
        dim3 g((Mpad + 127) / 128, (p.N + 63) / 64);
        wmma_conv_gemm<<<g, 256, 0, stream>>>(p);
      }
  };
  // BN: stats (partials) -> reduce -> apply
  auto bn = [&](const float* x, long long P, int C,
                const float* g, const float* b,
                __bf16* outb, float* outf, int relu) {
    int chunks = (int)(P / PPB);
    bn_stats_kernel<<<dim3(C, chunks), 256, 0, stream>>>(x, PSUM, PSQ, C, chunks);
    bn_reduce_kernel<<<C, 256, 0, stream>>>(PSUM, PSQ, SUMB, SQB, chunks);
    long long total = P * C;
    bn_apply_kernel<<<gsb(total), 256, 0, stream>>>(x, SUMB, SQB, g, b, outb, outf,
                                                    total, C, 1.0f / (float)P, relu);
  };

  // ================= pipeline =================
  // Input packing + weight reorders (bf16, im2col-friendly layouts)
  pack_image_kernel<<<gsb(16LL * 256 * 256 * 16), 256, 0, stream>>>(ori, XP, 256, 256);
  reorder_conv_w_kernel<<<gsb(512LL * 256), 256, 0, stream>>>(e_c1_w, W_C1, 512, 3, 16, 4, 4);
  reorder_conv_w_kernel<<<gsb(512LL * 8192), 256, 0, stream>>>(e_c2_w, W_C2, 512, 512, 512, 4, 4);
  reorder_conv_w_kernel<<<gsb(512LL * 4608), 256, 0, stream>>>(e_r1_w, W_ER1, 512, 512, 512, 3, 3);
  reorder_conv_w_kernel<<<gsb(512LL * 512), 256, 0, stream>>>(e_r2_w, W_ER2, 512, 512, 512, 1, 1);
  reorder_conv_w_kernel<<<gsb(512LL * 4608), 256, 0, stream>>>(d_r1_w, W_DR1, 512, 512, 512, 3, 3);
  reorder_conv_w_kernel<<<gsb(512LL * 512), 256, 0, stream>>>(d_r2_w, W_DR2, 512, 512, 512, 1, 1);
  cast_bf16_kernel<<<gsb(512LL * 512), 256, 0, stream>>>(emb, W_EMB, 512LL * 512);
  reorder_convt_w_kernel<<<gsb(4LL * 512 * 2048), 256, 0, stream>>>(d_t1_w, W_T1, 512, 512, 512);
  reorder_convt_w_kernel<<<gsb(4LL * 16 * 2048), 256, 0, stream>>>(d_t2_w, W_T2, 3, 16, 512);
  pad_bias_kernel<<<1, 256, 0, stream>>>(d_t2_b, BT2, 3, 16);

  // ---- Encoder ----
  launch_conv(W_C1, XP, BIG1, nullptr, 512, 256, 256, 16, 128, 128, 4, 4, 2, 1, 512);
  bn(BIG1, P128, 512, e_bn1_g, e_bn1_b, BIG2, nullptr, 1);
  launch_conv(W_C2, BIG2, M64_1, nullptr, 512, 128, 128, 512, 64, 64, 4, 4, 2, 1, 512);
  bn(M64_1, P64, 512, e_bn2_g, e_bn2_b, M64B1, M64_2, 1);          // X2 (bf16 + f32)
  launch_conv(W_ER1, M64B1, M64_3, nullptr, 512, 64, 64, 512, 64, 64, 3, 3, 1, 1, 512);
  bn(M64_3, P64, 512, e_rbn1_g, e_rbn1_b, M64B2, nullptr, 1);
  launch_conv(W_ER2, M64B2, M64_4, e_r2_b, 512, 64, 64, 512, 64, 64, 1, 1, 1, 0, 512);
  bn(M64_4, P64, 512, e_rbn2_g, e_rbn2_b, M64B2, M64_3, 1);        // r3 (f32 in M64_3)
  add_inplace_kernel<<<gsb(P64 * 512), 256, 0, stream>>>(M64_2, M64_3, P64 * 512);
  bn(M64_2, P64, 512, e_bn3_g, e_bn3_b, M64B3, nullptr, 0);        // z_e (bf16)

  // ---- Vector quantization ----
  launch_conv(W_EMB, M64B3, M64_1, nullptr, 512, 64, 64, 512, 64, 64, 1, 1, 1, 0, 512); // scores
  enorm_kernel<<<512, 256, 0, stream>>>(emb, ENORM, 512);
  argmin_kernel<<<(unsigned)P64, 256, 0, stream>>>(M64_1, ENORM, IDX, 512);
  gather_kernel<<<gsb(P64 * 512), 256, 0, stream>>>(emb, IDX, M64B1, M64_4, P64, 512); // z_q

  // ---- Decoder ----
  launch_conv(W_DR1, M64B1, M64_3, nullptr, 512, 64, 64, 512, 64, 64, 3, 3, 1, 1, 512);
  bn(M64_3, P64, 512, d_rbn1_g, d_rbn1_b, M64B2, nullptr, 1);
  launch_conv(W_DR2, M64B2, M64_1, nullptr, 512, 64, 64, 512, 64, 64, 1, 1, 1, 0, 512);
  bn(M64_1, P64, 512, d_rbn2_g, d_rbn2_b, M64B2, M64_3, 1);        // r3 (f32)
  add_inplace_kernel<<<gsb(P64 * 512), 256, 0, stream>>>(M64_4, M64_3, P64 * 512);
  bn(M64_4, P64, 512, d_bn0_g, d_bn0_b, M64B3, nullptr, 1);
  launch_convt(W_T1, M64B3, BIG1, nullptr, 512, 512, 64, 64, 512);
  bn(BIG1, P128, 512, d_bn1_g, d_bn1_b, BIG2, nullptr, 1);
  launch_convt(W_T2, BIG2, YF, BT2, 3, 16, 128, 128, 512);

  // ---- Final BN (C=3, channel stride 16) + tanh -> NCHW output ----
  {
    int chunks = (int)(P256 / PPB);
    bn_stats_kernel<<<dim3(3, chunks), 256, 0, stream>>>(YF, PSUM, PSQ, 16, chunks);
    bn_reduce_kernel<<<3, 256, 0, stream>>>(PSUM, PSQ, SUMB, SQB, chunks);
    final_out_kernel<<<gsb(16LL * 3 * 256 * 256), 256, 0, stream>>>(
        YF, SUMB, SQB, d_bn2_g, d_bn2_b, out, 256, 256, 1.0f / (float)P256);
  }
}
